// MLP_learner_4080218931472
// MI455X (gfx1250) — matrix operand order repaired and checked by reading, not compiled, not run on hardware
//
#include <hip/hip_runtime.h>
#include <hip/hip_bf16.h>

// ---------------------------------------------------------------------------
// Types for CDNA5 WMMA (wave32, 16x16x32 bf16 -> f32)
// ---------------------------------------------------------------------------
typedef __attribute__((ext_vector_type(16))) __bf16 v16bf;
typedef __attribute__((ext_vector_type(8)))  float  v8f;
typedef __attribute__((ext_vector_type(4)))  unsigned int v4u;

union BF16Tile {
    v16bf bf;
    v4u   u[2];   // two 16-byte halves
};

#define N_ROWS 16384
#define DIM    256
#define K_KEEP 31

// ---------------------------------------------------------------------------
// Kernel 1: 2-layer MLP + ReLU + L2 row-normalize -> bf16 Hn in workspace.
// One 256-thread block per row. FLOP-trivial (4 GF total); W1/W2 (256KB each)
// stay L2-resident across all 16384 blocks.
// ---------------------------------------------------------------------------
__global__ void mlp_norm_kernel(const float* __restrict__ X,
                                const float* __restrict__ W1,
                                const float* __restrict__ b1,
                                const float* __restrict__ W2,
                                const float* __restrict__ b2,
                                __hip_bfloat16* __restrict__ Hn) {
    __shared__ float xs[DIM];
    __shared__ float h1s[DIM];
    __shared__ float ssum;
    const int i = blockIdx.x;
    const int j = threadIdx.x;

    xs[j] = X[(size_t)i * DIM + j];
    if (j == 0) ssum = 0.0f;
    __syncthreads();

    // h1[j] = relu(dot(x_row, W1[j,:]) + b1[j])
    float acc = b1[j];
    const float4* w = (const float4*)(W1 + (size_t)j * DIM);
#pragma unroll 8
    for (int k = 0; k < DIM / 4; ++k) {
        float4 wv = w[k];
        acc += wv.x * xs[4 * k + 0] + wv.y * xs[4 * k + 1] +
               wv.z * xs[4 * k + 2] + wv.w * xs[4 * k + 3];
    }
    h1s[j] = fmaxf(acc, 0.0f);
    __syncthreads();

    // h2[j] = dot(h1_row, W2[j,:]) + b2[j]
    float acc2 = b2[j];
    const float4* w2 = (const float4*)(W2 + (size_t)j * DIM);
#pragma unroll 8
    for (int k = 0; k < DIM / 4; ++k) {
        float4 wv = w2[k];
        acc2 += wv.x * h1s[4 * k + 0] + wv.y * h1s[4 * k + 1] +
                wv.z * h1s[4 * k + 2] + wv.w * h1s[4 * k + 3];
    }

    atomicAdd(&ssum, acc2 * acc2);
    __syncthreads();

    const float inv = 1.0f / fmaxf(sqrtf(ssum), 1e-12f);
    Hn[(size_t)i * DIM + j] = __float2bfloat16(acc2 * inv);
}

// ---------------------------------------------------------------------------
// Kernel 2: sim = Hn @ Hn^T via v_wmma_f32_16x16x32_bf16.
// Block = 256 threads = 8 waves arranged 4(M) x 2(N); each wave owns a
// 32(M) x 64(N) output tile (2x4 WMMA accumulators -> 8 WMMAs per K-step
// against 12 b128 loads = 1.5 loads/WMMA). Workgroup tile = 128 x 128.
// B = Hn^T, so a B-operand column is a contiguous bf16 row of Hn; all
// operand traffic is L2-resident (Hn = 8 MB << 192 MB L2).
// ---------------------------------------------------------------------------
__global__ void sim_wmma_kernel(const __hip_bfloat16* __restrict__ Hn_,
                                float* __restrict__ out) {
    const __bf16* H = reinterpret_cast<const __bf16*>(Hn_);

    const int lane = threadIdx.x & 31;
    const int wave = threadIdx.x >> 5;           // 0..7
    const int wm   = wave & 3;                   // 4 waves along M
    const int wn   = wave >> 2;                  // 2 waves along N
    const int m0   = blockIdx.x * 128 + wm * 32;
    const int n0   = blockIdx.y * 128 + wn * 64;
    const int l15  = lane & 15;
    const int half = lane >> 4;                  // 0: lanes 0-15, 1: lanes 16-31

    v8f acc[2][4];
#pragma unroll
    for (int i = 0; i < 2; ++i)
#pragma unroll
        for (int j = 0; j < 4; ++j) acc[i][j] = (v8f){};

#pragma unroll
    for (int k0 = 0; k0 < DIM; k0 += 32) {
        // A tiles (16x32 bf16): lane<16 holds K k0..k0+15,
        // lane>=16 holds K k0+16..k0+31 of row m0+l15.
        BF16Tile a[2];
        const __bf16* pa = H + (size_t)(m0 + l15) * DIM + k0 + half * 16;
#pragma unroll
        for (int i = 0; i < 2; ++i) {
            a[i].u[0] = *(const v4u*)(pa + i * 16 * DIM);
            a[i].u[1] = *(const v4u*)(pa + i * 16 * DIM + 8);
        }
        // B tiles (32x16 bf16, B = Hn^T): lane<16 holds K k0..k0+15 of column
        // n0+l15 == contiguous 32 bytes of Hn row n0+l15; lane>=16 holds +16.
        BF16Tile b[4];
        const __bf16* pb = H + (size_t)(n0 + l15) * DIM + k0 + half * 16;
#pragma unroll
        for (int j = 0; j < 4; ++j)
            b[j].bf = *(const v16bf*)(pb + j * 16 * DIM);

#pragma unroll
        for (int i = 0; i < 2; ++i)
#pragma unroll
            for (int j = 0; j < 4; ++j)
                acc[i][j] = __builtin_amdgcn_wmma_f32_16x16x32_bf16(
                    false, a[i].bf, false, b[j].bf, (short)0, acc[i][j],
                    false, false);
    }

    // D layout: VGPR r -> row (r + 8*half), col = l15 within each 16x16 tile.
#pragma unroll
    for (int i = 0; i < 2; ++i) {
#pragma unroll
        for (int r = 0; r < 8; ++r) {
            const size_t row = (size_t)(m0 + i * 16 + half * 8 + r) * N_ROWS;
#pragma unroll
            for (int j = 0; j < 4; ++j)
                out[row + n0 + j * 16 + l15] = acc[i][j][r];
        }
    }
}

// ---------------------------------------------------------------------------
// Kernel 3: in-place per-row top-31 mask + ReLU on sim.
// One 256-thread block per row; each thread holds 64 values in registers.
// 30-step bisection on t in [-2,2] finds the 31st-largest threshold via
// count(x >= t); then rewrite with mask+relu (float4 stores).
// ---------------------------------------------------------------------------
__global__ void topk_relu_kernel(float* __restrict__ sim) {
    const int t = threadIdx.x;                    // 0..255
    float4* r4 = (float4*)(sim + (size_t)blockIdx.x * N_ROWS);

    float v[64];
#pragma unroll
    for (int c = 0; c < 16; ++c) {
        float4 x = r4[c * 256 + t];
        v[c * 4 + 0] = x.x; v[c * 4 + 1] = x.y;
        v[c * 4 + 2] = x.z; v[c * 4 + 3] = x.w;
    }

    __shared__ int s_cnt;
    float lo = -2.0f, hi = 2.0f;
    for (int it = 0; it < 30; ++it) {
        const float mid = 0.5f * (lo + hi);
        int c = 0;
#pragma unroll
        for (int e = 0; e < 64; ++e) c += (v[e] >= mid) ? 1 : 0;
        if (t == 0) s_cnt = 0;
        __syncthreads();
        atomicAdd(&s_cnt, c);
        __syncthreads();
        const int total = s_cnt;
        __syncthreads();
        if (total >= K_KEEP) lo = mid; else hi = mid;
    }
    const float thr = lo;

#pragma unroll
    for (int c = 0; c < 16; ++c) {
        float4 x;
        const float a = v[c * 4 + 0], b = v[c * 4 + 1];
        const float d = v[c * 4 + 2], e = v[c * 4 + 3];
        x.x = (a >= thr && a > 0.0f) ? a : 0.0f;
        x.y = (b >= thr && b > 0.0f) ? b : 0.0f;
        x.z = (d >= thr && d > 0.0f) ? d : 0.0f;
        x.w = (e >= thr && e > 0.0f) ? e : 0.0f;
        r4[c * 256 + t] = x;
    }
}

// ---------------------------------------------------------------------------
extern "C" void kernel_launch(void* const* d_in, const int* in_sizes, int n_in,
                              void* d_out, int out_size, void* d_ws, size_t ws_size,
                              hipStream_t stream) {
    const float* X  = (const float*)d_in[0];   // features [16384,256]
    const float* W1 = (const float*)d_in[1];   // [256,256]
    const float* b1 = (const float*)d_in[2];   // [256]
    const float* W2 = (const float*)d_in[3];   // [256,256]
    const float* b2 = (const float*)d_in[4];   // [256]
    float* out = (float*)d_out;                // [16384,16384]

    __hip_bfloat16* Hn = (__hip_bfloat16*)d_ws;  // 16384*256*2 = 8 MB

    // 1) MLP + L2-normalize -> bf16 Hn
    mlp_norm_kernel<<<dim3(N_ROWS), dim3(DIM), 0, stream>>>(X, W1, b1, W2, b2, Hn);

    // 2) sim = Hn @ Hn^T  (WMMA bf16 -> f32), 128x128 block tile
    dim3 grid(N_ROWS / 128, N_ROWS / 128);
    sim_wmma_kernel<<<grid, dim3(256), 0, stream>>>(Hn, out);

    // 3) per-row top-31 + relu, in place
    topk_relu_kernel<<<dim3(N_ROWS), dim3(256), 0, stream>>>(out);
}
